// SSIM_50190987821685
// MI455X (gfx1250) — matrix operand order repaired and checked by reading, not compiled, not run on hardware
//
#include <hip/hip_runtime.h>
#include <hip/hip_bf16.h>

typedef __attribute__((ext_vector_type(16))) _Float16 v16h;
typedef __attribute__((ext_vector_type(8)))  _Float16 v8h;
typedef __attribute__((ext_vector_type(8)))  float    v8f;
typedef __attribute__((ext_vector_type(4)))  float    v4f;

#define IMG 512
#define TILE 16
#define WAVES_PER_BLOCK 8
#define NPLANES 48                      // 16 batch * 3 channels
#define TILES_PER_DIM (IMG / TILE)      // 32
#define TILES_PER_PLANE (TILES_PER_DIM * TILES_PER_DIM)  // 1024
#define NTILES (NPLANES * TILES_PER_PLANE)               // 49152
#define NBLOCKS (NTILES / WAVES_PER_BLOCK)               // 6144

#define C1f 0.0001f
#define C2f 0.0009f

// ---- CDNA5 async Global->LDS path (guarded; falls back to plain loads) ----
#if defined(__has_builtin)
#if __has_builtin(__builtin_amdgcn_global_load_async_to_lds_b32)
#define HAVE_ASYNC_LDS 1
#endif
#if __has_builtin(__builtin_amdgcn_s_wait_asynccnt)
#define HAVE_WAIT_ASYNC_BUILTIN 1
#endif
#endif
#ifndef HAVE_ASYNC_LDS
#define HAVE_ASYNC_LDS 0
#endif
#ifndef HAVE_WAIT_ASYNC_BUILTIN
#define HAVE_WAIT_ASYNC_BUILTIN 0
#endif

// Compiler-only scheduling fence: per-wave LDS RAW needs no HW barrier
// (same-wave LDS ops are processed in order), only codegen ordering.
__device__ inline void sched_fence() { asm volatile("" ::: "memory"); }

#if HAVE_ASYNC_LDS
typedef __attribute__((address_space(1))) int gas_int_t;  // global
typedef __attribute__((address_space(3))) int las_int_t;  // LDS
__device__ inline void async_g2l_b32(const float* gsrc, float* ldst) {
    __builtin_amdgcn_global_load_async_to_lds_b32(
        (gas_int_t*)gsrc, (las_int_t*)ldst, 0, 0);
}
__device__ inline void wait_async0() {
#if HAVE_WAIT_ASYNC_BUILTIN
    __builtin_amdgcn_s_wait_asynccnt(0);
    asm volatile("" ::: "memory");
#else
    asm volatile("s_wait_asynccnt 0x0" ::: "memory");
#endif
}
#endif

// Normalized 11-tap Gaussian, sigma = 1.5 (matches reference _create_window)
__device__ __constant__ float G11[11] = {
    0.00102840f, 0.00759850f, 0.03600077f, 0.10936069f, 0.21300553f,
    0.26601174f,
    0.21300553f, 0.10936069f, 0.03600077f, 0.00759850f, 0.00102840f};

__device__ inline v8f wmma_f16(v16h a, v16h b, v8f c) {
    // (neg_a, A, neg_b, B, c_mod, C, reuse_a, reuse_b)
    return __builtin_amdgcn_wmma_f32_16x16x32_f16(false, a, false, b,
                                                  (short)0, c, false, false);
}

// 16-bit A-matrix 16x32 layout (ISA 7.12.2):
// lanes 0-15  hold row M=lane,    elems: K=0..7, K=16..23
// lanes 16-31 hold row M=lane-16, elems: K=8..15, K=24..31
// Region is f32 in LDS; convert to f16 while building the fragment.
__device__ inline v16h make_a_frag_f32(const float* region /*32x32 row-major*/,
                                       int lane, int rbase) {
    const int M    = lane & 15;
    const int koff = (lane < 16) ? 0 : 8;
    const float* row = region + (rbase + M) * 32;
    v4f a0 = *(const v4f*)(row + koff);
    v4f a1 = *(const v4f*)(row + koff + 4);
    v4f b0 = *(const v4f*)(row + 16 + koff);
    v4f b1 = *(const v4f*)(row + 16 + koff + 4);
    v16h a;
#pragma unroll
    for (int e = 0; e < 4; ++e) {
        a[e]      = (_Float16)a0[e];
        a[4 + e]  = (_Float16)a1[e];
        a[8 + e]  = (_Float16)b0[e];
        a[12 + e] = (_Float16)b1[e];
    }
    return a;
}

// 16-bit B-matrix 32x16 layout: lane = column N (lanes 0-15: K=0..7,16..23,
// lanes 16-31: K=8..15,24..31), matching the A-fragment K order.
// Horizontal Gaussian band: Gh[t][j] = g[t-j], valid 0 <= t-j <= 10, t < 26.
__device__ inline v16h make_gh_bfrag(int lane) {
    const int n  = lane & 15;
    const int kb = (lane < 16) ? 0 : 8;
    v16h b;
#pragma unroll
    for (int e = 0; e < 16; ++e) {
        int t = (e < 8) ? (kb + e) : (16 + kb + (e - 8));
        int d = t - n;
        float v = (t < 26 && d >= 0 && d <= 10) ? G11[d] : 0.0f;
        b[e] = (_Float16)v;
    }
    return b;
}

// Vertical Gaussian band as A-matrix: Gv[i][t] = g[t-i].
__device__ inline v16h make_gv_afrag(int lane) {
    const int M    = lane & 15;
    const int koff = (lane < 16) ? 0 : 8;
    v16h a;
#pragma unroll
    for (int e = 0; e < 16; ++e) {
        int K = (e < 8) ? (koff + e) : (16 + koff + (e - 8));
        int d = K - M;
        float v = (K < 26 && d >= 0 && d <= 10) ? G11[d] : 0.0f;
        a[e] = (_Float16)v;
    }
    return a;
}

__device__ inline v16h vmul16(v16h a, v16h b) {
    v16h r;
#pragma unroll
    for (int e = 0; e < 16; ++e) r[e] = a[e] * b[e];
    return r;
}

__global__ __launch_bounds__(256) void ssim_tile_kernel(
    const float* __restrict__ img1, const float* __restrict__ img2,
    float* __restrict__ partial) {
    // Per-wave staging buffers (waves are fully independent)
    __shared__ __align__(16) float    sXf[WAVES_PER_BLOCK][32][32];
    __shared__ __align__(16) float    sYf[WAVES_PER_BLOCK][32][32];
    __shared__ __align__(16) _Float16 sH[WAVES_PER_BLOCK][16][32]; // [col][row]
    __shared__ float wsum[WAVES_PER_BLOCK];

    const int lane = threadIdx.x & 31;
    const int wave = threadIdx.x >> 5;

    const int tileId = blockIdx.x * WAVES_PER_BLOCK + wave;
    const int plane  = tileId >> 10;          // / TILES_PER_PLANE
    const int tp     = tileId & 1023;
    const int ty     = tp >> 5;
    const int tx     = tp & 31;
    const int row0   = ty * TILE;
    const int col0   = tx * TILE;

    const float* p1 = img1 + (size_t)plane * (IMG * IMG);
    const float* p2 = img2 + (size_t)plane * (IMG * IMG);

    // ---- Stage 26x26 halo region (zero-padded to 32x32) as f32 in LDS ----
    // Valid elements go memory->LDS via the async path (ASYNCcnt); padding
    // zeros go via DS stores. The two touch disjoint addresses, so the only
    // ordering needed is s_wait_asynccnt before the DS fragment reads.
#pragma unroll 8
    for (int r = 0; r < 32; ++r) {
        const int gr = row0 - 5 + r;
        const int gc = col0 - 5 + lane;
        const bool ok = (r < 26) && (lane < 26) &&
                        (gr >= 0) && (gr < IMG) && (gc >= 0) && (gc < IMG);
#if HAVE_ASYNC_LDS
        if (ok) {
            const int idx = gr * IMG + gc;
            async_g2l_b32(p1 + idx, &sXf[wave][r][lane]);
            async_g2l_b32(p2 + idx, &sYf[wave][r][lane]);
        } else {
            sXf[wave][r][lane] = 0.0f;
            sYf[wave][r][lane] = 0.0f;
        }
#else
        float x = 0.0f, y = 0.0f;
        if (ok) {
            const int idx = gr * IMG + gc;
            x = p1[idx];
            y = p2[idx];
        }
        sXf[wave][r][lane] = x;
        sYf[wave][r][lane] = y;
#endif
    }
#if HAVE_ASYNC_LDS
    wait_async0();
#endif
    sched_fence();

    // Constant operand fragments (reused for all 5 planes)
    const v16h bGh = make_gh_bfrag(lane);
    const v16h aGv = make_gv_afrag(lane);

    // Input A-fragments for both row-halves of the region (f32 -> f16)
    const v16h ax0 = make_a_frag_f32(&sXf[wave][0][0], lane, 0);
    const v16h ax1 = make_a_frag_f32(&sXf[wave][0][0], lane, 16);
    const v16h ay0 = make_a_frag_f32(&sYf[wave][0][0], lane, 0);
    const v16h ay1 = make_a_frag_f32(&sYf[wave][0][0], lane, 16);

    const int N   = lane & 15;
    const int mb  = (lane < 16) ? 0 : 8;   // C/D row base within lane half
    const int kb2 = (lane < 16) ? 0 : 8;   // B-frag K base

    v8f res[5];
    const v8f zero = {};

#pragma unroll
    for (int p = 0; p < 5; ++p) {
        v16h a0, a1;
        if (p == 0)      { a0 = ax0;              a1 = ax1; }
        else if (p == 1) { a0 = ay0;              a1 = ay1; }
        else if (p == 2) { a0 = vmul16(ax0, ax0); a1 = vmul16(ax1, ax1); }
        else if (p == 3) { a0 = vmul16(ay0, ay0); a1 = vmul16(ay1, ay1); }
        else             { a0 = vmul16(ax0, ay0); a1 = vmul16(ax1, ay1); }

        // Horizontal pass: H(32x16) = region(32x26) x Gh(26x16), two WMMAs
        v8f h0 = wmma_f16(a0, bGh, zero);   // H rows 0..15
        v8f h1 = wmma_f16(a1, bGh, zero);   // H rows 16..31

        // Repackage H (C/D layout, f32) -> column-major f16 in LDS.
        // Same-wave LDS ops execute in order: only a compiler fence needed.
#pragma unroll
        for (int i = 0; i < 8; ++i) {
            sH[wave][N][mb + i]      = (_Float16)h0[i];
            sH[wave][N][16 + mb + i] = (_Float16)h1[i];
        }
        sched_fence();

        // Reload H as B-fragment (32x16)
        v8h blo = *(const v8h*)&sH[wave][N][kb2];
        v8h bhi = *(const v8h*)&sH[wave][N][kb2 + 16];
        v16h bh;
#pragma unroll
        for (int e = 0; e < 8; ++e) { bh[e] = blo[e]; bh[8 + e] = bhi[e]; }

        // Vertical pass: Out(16x16) = Gv(16x26) x H(26x16)
        res[p] = wmma_f16(aGv, bh, zero);
        sched_fence();  // order bh loads before next plane's sH stores
    }

    // ---- SSIM map + local reduction (all frags share element->(M,N) map) ----
    float acc = 0.0f;
#pragma unroll
    for (int i = 0; i < 8; ++i) {
        const float mu1 = res[0][i];
        const float mu2 = res[1][i];
        const float exx = res[2][i];
        const float eyy = res[3][i];
        const float exy = res[4][i];
        const float mu1s = mu1 * mu1;
        const float mu2s = mu2 * mu2;
        const float m12  = mu1 * mu2;
        const float s1   = exx - mu1s;
        const float s2   = eyy - mu2s;
        const float s12  = exy - m12;
        const float num  = (2.0f * m12 + C1f) * (2.0f * s12 + C2f);
        const float den  = (mu1s + mu2s + C1f) * (s1 + s2 + C2f);
        acc += num / den;
    }

    // wave32 tree reduction
#pragma unroll
    for (int off = 16; off > 0; off >>= 1)
        acc += __shfl_xor(acc, off, 32);

    if (lane == 0) wsum[wave] = acc;
    __syncthreads();   // only cross-wave interaction in the kernel
    if (threadIdx.x == 0) {
        float s = 0.0f;
#pragma unroll
        for (int w = 0; w < WAVES_PER_BLOCK; ++w) s += wsum[w];
        partial[blockIdx.x] = s;
    }
}

__global__ __launch_bounds__(256) void ssim_reduce_kernel(
    const float* __restrict__ partial, float* __restrict__ out) {
    __shared__ float sh[256];
    float s = 0.0f;
    for (int i = threadIdx.x; i < NBLOCKS; i += 256) s += partial[i];
    sh[threadIdx.x] = s;
    __syncthreads();
#pragma unroll
    for (int st = 128; st > 0; st >>= 1) {
        if (threadIdx.x < st) sh[threadIdx.x] += sh[threadIdx.x + st];
        __syncthreads();
    }
    if (threadIdx.x == 0) {
        const float invN = 1.0f / (float)(16 * 3 * IMG * IMG);
        out[0] = sh[0] * invN;
    }
}

extern "C" void kernel_launch(void* const* d_in, const int* in_sizes, int n_in,
                              void* d_out, int out_size, void* d_ws, size_t ws_size,
                              hipStream_t stream) {
    const float* img1 = (const float*)d_in[0];
    const float* img2 = (const float*)d_in[1];
    // d_in[2] is the Gaussian window; we use exact hardcoded constants instead.
    float* partial = (float*)d_ws;           // NBLOCKS floats of scratch
    float* out     = (float*)d_out;

    ssim_tile_kernel<<<NBLOCKS, 256, 0, stream>>>(img1, img2, partial);
    ssim_reduce_kernel<<<1, 256, 0, stream>>>(partial, out);
}
